// WindowAttention_37417755083132
// MI455X (gfx1250) — matrix operand order repaired and checked by reading, not compiled, not run on hardware
//
#include <hip/hip_runtime.h>
#include <cstdint>

typedef unsigned short u16;
typedef __attribute__((ext_vector_type(16))) __bf16 v16bf;
typedef __attribute__((ext_vector_type(16))) u16    v16u;
typedef __attribute__((ext_vector_type(8)))  u16    v8u;
typedef __attribute__((ext_vector_type(8)))  float  v8f;

#define NWIN 4096
#define NTOK 49
#define CDIM 192
#define NH   6
#define HD   32
#define NWMASK 64
#define SCALE_Q 0.17677669529663687f

// ---------- helpers ----------
__device__ __forceinline__ u16 f2bfu(float f) {
  unsigned u = __builtin_bit_cast(unsigned, f);
  u += 0x7fffu + ((u >> 16) & 1u);      // round-to-nearest-even
  return (u16)(u >> 16);
}

// A-fragment (16x16x32 bf16): lane<16 holds K 0..15 (klo=0), lane>=16 holds K 16..31 (klo=8).
// rowk points at (row, K-window base).
__device__ __forceinline__ v16bf afrag(const u16* rowk, int klo) {
  v8u lo = *(const v8u*)(rowk + 2 * klo);
  v8u hi = *(const v8u*)(rowk + 2 * klo + 8);
  v16u r;
#pragma unroll
  for (int i = 0; i < 8; ++i) { r[i] = lo[i]; r[i + 8] = hi[i]; }
  return __builtin_bit_cast(v16bf, r);
}
// B-fragment: 16 contiguous K halves per lane (lane<16 -> K 0..15, lane>=16 -> K 16..31).
__device__ __forceinline__ v16bf bfrag(const u16* p) {
  return __builtin_bit_cast(v16bf, *(const v16u*)p);
}
__device__ __forceinline__ v16bf zfrag() {
  v16u z;
#pragma unroll
  for (int i = 0; i < 16; ++i) z[i] = 0;
  return __builtin_bit_cast(v16bf, z);
}
__device__ __forceinline__ v8f wmma_bf(v16bf a, v16bf b, v8f c) {
  return __builtin_amdgcn_wmma_f32_16x16x32_bf16(false, a, false, b, (short)0, c, false, false);
}

// ---------- kernel 0: weight transpose->bf16 + rpb gather ----------
__global__ __launch_bounds__(256) void prep_kernel(
    const float* __restrict__ Wqkv, const float* __restrict__ Wproj,
    const float* __restrict__ rpb_t,
    u16* __restrict__ wqkvT, u16* __restrict__ wprojT, float* __restrict__ rpbf) {
  int i = blockIdx.x * 256 + threadIdx.x;
  if (i < 576 * 192) {                       // wqkvT[c][k] = Wqkv[k][c]
    int c = i / 192, k = i - c * 192;
    wqkvT[i] = f2bfu(Wqkv[k * 576 + c]);
  } else if (i < 576 * 192 + 192 * 192) {    // wprojT[c][k] = Wproj[k][c]
    int j = i - 576 * 192;
    int c = j / 192, k = j - c * 192;
    wprojT[j] = f2bfu(Wproj[k * 192 + c]);
  } else if (i < 576 * 192 + 192 * 192 + NH * NTOK * NTOK) {
    int j = i - (576 * 192 + 192 * 192);
    int h = j / (NTOK * NTOK), rc = j - h * NTOK * NTOK;
    int r = rc / NTOK, c = rc - r * NTOK;
    int ri = r / 7, rj = r - ri * 7, ci = c / 7, cj = c - ci * 7;
    int idx = (ri - ci + 6) * 13 + (rj - cj + 6);
    rpbf[j] = rpb_t[idx * NH + h];
  }
}

// ---------- kernel 1: zero padded rows of K (tokens 49..63) and V^T (keys 49..63) ----------
__global__ __launch_bounds__(256) void pad_kernel(u16* __restrict__ kbuf, u16* __restrict__ vT) {
  const int PK = NWIN * NH * 15 * 32;        // 11,796,480
  int i = blockIdx.x * 256 + threadIdx.x;
  if (i < PK) {
    int bh = i / 480; int rem = i - bh * 480;
    int n = 49 + rem / 32, d = rem & 31;
    kbuf[((size_t)bh * 64 + n) * 32 + d] = 0;
  } else if (i < 2 * PK) {
    int j = i - PK;
    int bh = j / 480; int rem = j - bh * 480;
    int d = rem / 15, key = 49 + rem % 15;
    vT[(size_t)bh * 2048 + d * 64 + key] = 0;
  }
}

// ---------- kernel 2: QKV GEMM (200704x576 = x(200704x192) @ Wqkv) ----------
__global__ __launch_bounds__(256) void qkv_gemm_kernel(
    const float* __restrict__ x, const float* __restrict__ bqkv,
    const u16* __restrict__ wqkvT,
    u16* __restrict__ qbuf, u16* __restrict__ kbuf, u16* __restrict__ vT) {
  __shared__ u16 a_s[128 * 32];
  __shared__ u16 b_s[64 * 32];
  const int t = threadIdx.x;
  const int m0 = blockIdx.x * 128, n0 = blockIdx.y * 64;
  const int lane = t & 31, wid = t >> 5, lh = lane & 15;
  const bool hih = lane >= 16;
  const int waveM = wid >> 1, waveN = wid & 1;
  const int klo = hih ? 8 : 0, kb16 = hih ? 16 : 0;

  v8f acc[2][2];
#pragma unroll
  for (int a = 0; a < 2; ++a)
#pragma unroll
    for (int bb = 0; bb < 2; ++bb)
#pragma unroll
      for (int v = 0; v < 8; ++v) acc[a][bb][v] = 0.0f;

  for (int ks = 0; ks < 6; ++ks) {
    const int kk = ks * 32;
    __syncthreads();
    {  // stage A: 128 rows x 32 cols of x as bf16
      int row = t >> 1, cb = (t & 1) * 16;
      const float4* xp = (const float4*)(x + (size_t)(m0 + row) * 192 + kk + cb);
#pragma unroll
      for (int i = 0; i < 4; ++i) {
        float4 v = xp[i];
        int o = row * 32 + cb + i * 4;
        a_s[o + 0] = f2bfu(v.x); a_s[o + 1] = f2bfu(v.y);
        a_s[o + 2] = f2bfu(v.z); a_s[o + 3] = f2bfu(v.w);
      }
    }
    {  // stage B: 64 cols x 32 k of wqkvT (already bf16, [col][k])
      int c = t >> 2, kb = (t & 3) * 8;
      *(v8u*)(b_s + c * 32 + kb) = *(const v8u*)(wqkvT + (size_t)(n0 + c) * 192 + kk + kb);
    }
    __syncthreads();
    v16bf af[2], bf[2];
#pragma unroll
    for (int ti = 0; ti < 2; ++ti)
      af[ti] = afrag(a_s + (waveM * 32 + ti * 16 + lh) * 32, klo);
#pragma unroll
    for (int tj = 0; tj < 2; ++tj)
      bf[tj] = bfrag(b_s + (waveN * 32 + tj * 16 + lh) * 32 + kb16);
#pragma unroll
    for (int ti = 0; ti < 2; ++ti)
#pragma unroll
      for (int tj = 0; tj < 2; ++tj)
        acc[ti][tj] = wmma_bf(af[ti], bf[tj], acc[ti][tj]);
  }

  // epilogue: + bias, scatter to q (scaled), k (row-stride 64), v^T
#pragma unroll
  for (int ti = 0; ti < 2; ++ti)
#pragma unroll
    for (int tj = 0; tj < 2; ++tj)
#pragma unroll
      for (int vr = 0; vr < 8; ++vr) {
        int row_g = m0 + waveM * 32 + ti * 16 + vr + (hih ? 8 : 0);
        int col_g = n0 + waveN * 32 + tj * 16 + lh;
        float val = acc[ti][tj][vr] + bqkv[col_g];
        int three = col_g / 192, rem = col_g - three * 192;
        int h = rem >> 5, d = rem & 31;
        int b = row_g / 49, n = row_g - b * 49;
        size_t bh = (size_t)b * NH + h;
        if (three == 0)      qbuf[(bh * 49 + n) * 32 + d] = f2bfu(val * SCALE_Q);
        else if (three == 1) kbuf[(bh * 64 + n) * 32 + d] = f2bfu(val);
        else                 vT[bh * 2048 + d * 64 + n]   = f2bfu(val);
      }
}

// ---------- kernel 3: fused attention + softmax + projection (one block per window) ----------
__global__ __launch_bounds__(256) void attn_kernel(
    const u16* __restrict__ qbuf, const u16* __restrict__ kbuf, const u16* __restrict__ vT,
    const float* __restrict__ rpbf, const float* __restrict__ mask,
    const float* __restrict__ M_win, const u16* __restrict__ wprojT_g,
    const float* __restrict__ bproj, float* __restrict__ out) {
  extern __shared__ char smem[];
  u16*   wt_s   = (u16*)smem;                 // 192*192       = 36864 halves
  u16*   p_s    = wt_s + 36864;               // 6*64*64       = 24576 halves
  u16*   ctx_s  = p_s + 24576;                // 64*192        = 12288 halves
  float* bias_s = (float*)(ctx_s + 12288);    // 49*49 floats

  const int t = threadIdx.x, b = blockIdx.x;
  const int lane = t & 31, wid = t >> 5, lh = lane & 15;
  const bool hih = lane >= 16;
  const int klo = hih ? 8 : 0, kb16 = hih ? 16 : 0;

  // stage W_proj^T and combined (mask + prior) bias
  for (int i = t; i < 36864 / 2; i += 256)
    ((unsigned*)wt_s)[i] = ((const unsigned*)wprojT_g)[i];
  const int w = b & (NWMASK - 1);
  for (int i = t; i < NTOK * NTOK; i += 256) {
    int r = i / NTOK, c = i - r * NTOK;
    float mr = fminf(fmaxf(M_win[b * NTOK + r], 0.f), 1.f);
    float mc = fminf(fmaxf(M_win[b * NTOK + c], 0.f), 1.f);
    bias_s[i] = mask[(size_t)w * NTOK * NTOK + i] + 0.5f * (mr + mc);
  }
  __syncthreads();

  if (wid < NH) {
    const int h = wid;
    const size_t bh = (size_t)b * NH + h;
    // ---- S = Q K^T (64x64 padded) ----
    v16bf qa[4], kf[4];
#pragma unroll
    for (int mt = 0; mt < 4; ++mt) {
      int row = mt * 16 + lh;
      qa[mt] = (row < NTOK) ? afrag(qbuf + (bh * 49 + row) * 32, klo) : zfrag();
    }
#pragma unroll
    for (int nt = 0; nt < 4; ++nt) {
      int col = nt * 16 + lh;
      kf[nt] = bfrag(kbuf + (bh * 64 + col) * 32 + kb16);
    }
    v8f acc[4][4];
#pragma unroll
    for (int mt = 0; mt < 4; ++mt)
#pragma unroll
      for (int nt = 0; nt < 4; ++nt) {
#pragma unroll
        for (int v = 0; v < 8; ++v) acc[mt][nt][v] = 0.0f;
        acc[mt][nt] = wmma_bf(qa[mt], kf[nt], acc[mt][nt]);
      }
    // ---- biases + mask ----
    const float* rpb_h = rpbf + h * NTOK * NTOK;
#pragma unroll
    for (int mt = 0; mt < 4; ++mt)
#pragma unroll
      for (int nt = 0; nt < 4; ++nt)
#pragma unroll
        for (int vr = 0; vr < 8; ++vr) {
          int row = mt * 16 + vr + (hih ? 8 : 0);
          int col = nt * 16 + lh;
          float s = -1e30f;
          if (row < NTOK && col < NTOK)
            s = acc[mt][nt][vr] + rpb_h[row * NTOK + col] + bias_s[row * NTOK + col];
          acc[mt][nt][vr] = s;
        }
    // ---- softmax over cols (16-lane half-wave holds one row across lanes) ----
#pragma unroll
    for (int mt = 0; mt < 4; ++mt)
#pragma unroll
      for (int vr = 0; vr < 8; ++vr) {
        float m = fmaxf(fmaxf(acc[mt][0][vr], acc[mt][1][vr]),
                        fmaxf(acc[mt][2][vr], acc[mt][3][vr]));
        m = fmaxf(m, __shfl_xor(m, 1, 32));
        m = fmaxf(m, __shfl_xor(m, 2, 32));
        m = fmaxf(m, __shfl_xor(m, 4, 32));
        m = fmaxf(m, __shfl_xor(m, 8, 32));
        float sum = 0.f;
#pragma unroll
        for (int nt = 0; nt < 4; ++nt) {
          float p = __expf(acc[mt][nt][vr] - m);
          acc[mt][nt][vr] = p;
          sum += p;
        }
        sum += __shfl_xor(sum, 1, 32);
        sum += __shfl_xor(sum, 2, 32);
        sum += __shfl_xor(sum, 4, 32);
        sum += __shfl_xor(sum, 8, 32);
        float r = 1.0f / sum;
#pragma unroll
        for (int nt = 0; nt < 4; ++nt) acc[mt][nt][vr] *= r;
      }
    // ---- P -> LDS (bf16, row-major stride 64) ----
    u16* ph = p_s + h * 4096;
#pragma unroll
    for (int mt = 0; mt < 4; ++mt)
#pragma unroll
      for (int nt = 0; nt < 4; ++nt)
#pragma unroll
        for (int vr = 0; vr < 8; ++vr) {
          int row = mt * 16 + vr + (hih ? 8 : 0);
          int col = nt * 16 + lh;
          ph[row * 64 + col] = f2bfu(acc[mt][nt][vr]);
        }
    // ---- O = P V (64x32) ----
    v8f oacc[4][2];
#pragma unroll
    for (int mt = 0; mt < 4; ++mt)
#pragma unroll
      for (int dt = 0; dt < 2; ++dt)
#pragma unroll
        for (int v = 0; v < 8; ++v) oacc[mt][dt][v] = 0.0f;
#pragma unroll
    for (int ksp = 0; ksp < 2; ++ksp) {
      v16bf pa[4], vb[2];
#pragma unroll
      for (int mt = 0; mt < 4; ++mt)
        pa[mt] = afrag(ph + (mt * 16 + lh) * 64 + ksp * 32, klo);
#pragma unroll
      for (int dt = 0; dt < 2; ++dt)
        vb[dt] = bfrag(vT + bh * 2048 + (dt * 16 + lh) * 64 + ksp * 32 + kb16);
#pragma unroll
      for (int mt = 0; mt < 4; ++mt)
#pragma unroll
        for (int dt = 0; dt < 2; ++dt)
          oacc[mt][dt] = wmma_bf(pa[mt], vb[dt], oacc[mt][dt]);
    }
    // ---- context -> LDS (64 x 192 bf16) ----
#pragma unroll
    for (int mt = 0; mt < 4; ++mt)
#pragma unroll
      for (int dt = 0; dt < 2; ++dt)
#pragma unroll
        for (int vr = 0; vr < 8; ++vr) {
          int row = mt * 16 + vr + (hih ? 8 : 0);
          int d = dt * 16 + lh;
          ctx_s[row * 192 + h * 32 + d] = f2bfu(oacc[mt][dt][vr]);
        }
  }
  __syncthreads();

  // ---- projection: out(64x192) = ctx(64x192) @ Wproj; 8 waves x (1 mtile, 6 ntiles) ----
  {
    const int mt = wid >> 1;
    const int jc0 = (wid & 1) * 6;
    v8f pacc[6];
#pragma unroll
    for (int j = 0; j < 6; ++j)
#pragma unroll
      for (int v = 0; v < 8; ++v) pacc[j][v] = 0.0f;
#pragma unroll
    for (int ks = 0; ks < 6; ++ks) {
      v16bf ca = afrag(ctx_s + (mt * 16 + lh) * 192 + ks * 32, klo);
#pragma unroll
      for (int j = 0; j < 6; ++j) {
        int colc = (jc0 + j) * 16 + lh;
        v16bf wb = bfrag(wt_s + colc * 192 + ks * 32 + kb16);
        pacc[j] = wmma_bf(ca, wb, pacc[j]);
      }
    }
#pragma unroll
    for (int j = 0; j < 6; ++j)
#pragma unroll
      for (int vr = 0; vr < 8; ++vr) {
        int row = mt * 16 + vr + (hih ? 8 : 0);
        if (row < NTOK) {
          int col = (jc0 + j) * 16 + lh;
          out[((size_t)b * NTOK + row) * CDIM + col] = pacc[j][vr] + bproj[col];
        }
      }
  }
}

// ---------- host ----------
extern "C" void kernel_launch(void* const* d_in, const int* in_sizes, int n_in,
                              void* d_out, int out_size, void* d_ws, size_t ws_size,
                              hipStream_t stream) {
  const float* x       = (const float*)d_in[0];
  const float* mask    = (const float*)d_in[1];
  const float* M_win   = (const float*)d_in[2];
  const float* W_qkv   = (const float*)d_in[3];
  const float* b_qkv   = (const float*)d_in[4];
  const float* rpb_t   = (const float*)d_in[5];
  const float* W_proj  = (const float*)d_in[6];
  const float* b_proj  = (const float*)d_in[7];
  float* out = (float*)d_out;
  char*  ws  = (char*)d_ws;

  u16*   wqkvT  = (u16*)(ws);                                   // 221,184 B
  u16*   wprojT = (u16*)(ws + 221184);                          //  73,728 B
  float* rpbf   = (float*)(ws + 294912);                        //  57,624 B (pad 57,856)
  u16*   qbuf   = (u16*)(ws + 352768);                          //  77,070,336 B
  u16*   kbuf   = (u16*)(ws + 352768 + 77070336ULL);            // 100,663,296 B
  u16*   vT     = (u16*)(ws + 352768 + 77070336ULL + 100663296ULL);

  prep_kernel<<<633, 256, 0, stream>>>(W_qkv, W_proj, rpb_t, wqkvT, wprojT, rpbf);
  pad_kernel<<<92160, 256, 0, stream>>>(kbuf, vT);
  qkv_gemm_kernel<<<dim3(1568, 9), 256, 0, stream>>>(x, b_qkv, wqkvT, qbuf, kbuf, vT);
  const size_t smem = (36864 + 24576 + 12288) * 2 + 2432 * 4;   // 157,184 B
  attn_kernel<<<NWIN, 256, smem, stream>>>(qbuf, kbuf, vT, rpbf, mask, M_win,
                                           wprojT, b_proj, out);
}
